// GRU_14413910245611
// MI455X (gfx1250) — matrix operand order repaired and checked by reading, not compiled, not run on hardware
//
#include <hip/hip_runtime.h>

typedef __attribute__((ext_vector_type(16))) _Float16 v16h;
typedef __attribute__((ext_vector_type(8)))  _Float16 v8h;
typedef __attribute__((ext_vector_type(8)))  float    v8f;

#define S_LEN  2048
#define BATCH_ 256
#define HDIM   512
#define DDIM   128
#define KTOT   640      // D + H
#define NBLK   32
#define TPB    256

#define WP  648         // lds_w row stride (halves): 1296B, 16B aligned, bank step 4
#define WOP 520         // lds_wo row stride (halves): 1040B
#define AP  40          // lds_a row stride (halves): 80B, bank step 20 (conflict-free)
#define ABUF (BATCH_ * AP)   // halves per A buffer

__device__ __forceinline__ v8f splat8(float s) {
  v8f r;
#pragma unroll
  for (int i = 0; i < 8; ++i) r[i] = s;
  return r;
}

__device__ __forceinline__ float sigf(float x) {
  return 1.0f / (1.0f + __expf(-x));
}

__device__ __forceinline__ v8f wmma_f16(v16h a, v16h b, v8f c) {
  // D = A(16x32 f16) * B(32x16 f16) + C(f32)
  return __builtin_amdgcn_wmma_f32_16x16x32_f16(false, a, false, b, (short)0, c,
                                                false, false);
}

__device__ __forceinline__ void wait_async0() {
#if __has_builtin(__builtin_amdgcn_s_wait_asynccnt)
  __builtin_amdgcn_s_wait_asynccnt(0);
#else
  asm volatile("s_wait_asynccnt 0" ::: "memory");
#endif
}

// Async DMA of 64B (4 x b128) from global f16 to LDS, per lane.
// INST_OFFSET is added to BOTH the LDS and global address (ISA 08 §4.4).
__device__ __forceinline__ void async_copy64(unsigned laddr,
                                             const _Float16* gsrc) {
  asm volatile("global_load_async_to_lds_b128 %0, %1, off"
               ::"v"(laddr), "v"(gsrc) : "memory");
  asm volatile("global_load_async_to_lds_b128 %0, %1, off offset:16"
               ::"v"(laddr), "v"(gsrc) : "memory");
  asm volatile("global_load_async_to_lds_b128 %0, %1, off offset:32"
               ::"v"(laddr), "v"(gsrc) : "memory");
  asm volatile("global_load_async_to_lds_b128 %0, %1, off offset:48"
               ::"v"(laddr), "v"(gsrc) : "memory");
}

__device__ __forceinline__ void grid_sync(unsigned* cnt, unsigned expected) {
  __threadfence();
  __syncthreads();
  if (threadIdx.x == 0) {
    atomicAdd(cnt, 1u);
    while (__hip_atomic_load(cnt, __ATOMIC_RELAXED, __HIP_MEMORY_SCOPE_AGENT) <
           expected) {
      __builtin_amdgcn_s_sleep(2);
    }
  }
  __syncthreads();
  __threadfence();
}

// Pack weights to f16 workspace: wcat[n][k] = k<128 ? w_ih[n][k] : w_hh[n][k-128]
__global__ __launch_bounds__(TPB) void gru_prep(
    const float* __restrict__ w_ih, const float* __restrict__ w_hh,
    const float* __restrict__ w_o, const float* __restrict__ h_init,
    _Float16* __restrict__ wcat, _Float16* __restrict__ wo16,
    _Float16* __restrict__ h0) {
  const int idx = blockIdx.x * TPB + threadIdx.x;
  if (idx < 3 * HDIM * KTOT) {
    const int n = idx / KTOT, k = idx % KTOT;
    const float v = (k < DDIM) ? w_ih[(size_t)n * DDIM + k]
                               : w_hh[(size_t)n * HDIM + (k - DDIM)];
    wcat[idx] = (_Float16)v;
  }
  if (idx < DDIM * HDIM) wo16[idx] = (_Float16)w_o[idx];
  if (idx < BATCH_ * HDIM) h0[idx] = (_Float16)h_init[idx];
}

__global__ __launch_bounds__(TPB) void gru_persistent(
    const float* __restrict__ x, const float* __restrict__ b_ih,
    const float* __restrict__ b_hh, const float* __restrict__ b_o,
    const _Float16* __restrict__ wcat, const _Float16* __restrict__ wo16,
    _Float16* __restrict__ hb0, _Float16* __restrict__ hb1,
    float* __restrict__ out, unsigned* __restrict__ cnt) {
  __shared__ _Float16 lds_w[48 * WP];      // 3 gates x 16 cols, K=640
  __shared__ _Float16 lds_wo[16 * WOP];    // 16 err cols, K=512
  __shared__ _Float16 lds_a[2 * ABUF];     // A chunk, double buffered

  const int tid  = threadIdx.x;
  const int w    = tid >> 5;      // wave 0..7
  const int l    = tid & 31;
  const int ln   = l & 15;
  const int hifl = l >> 4;        // 0 or 1
  const int c0   = blockIdx.x * 16;        // this block's h columns
  const int n0   = (blockIdx.x & 7) * 16;  // this block's err columns
  const int mgrp = blockIdx.x >> 3;        // err M-tile group

  // ---- Stage gate weights into LDS once (48 rows x 640 halves) ----
  for (int u = tid; u < 48 * 40; u += TPB) {
    const int r = u / 40, c = (u % 40) * 16;
    const int g = r >> 4, j = r & 15;
    const uint4* s =
        (const uint4*)(wcat + (size_t)(g * HDIM + c0 + j) * KTOT + c);
    uint4* dv = (uint4*)&lds_w[r * WP + c];
    dv[0] = s[0];
    dv[1] = s[1];
  }
  // ---- Stage output-projection weights (16 rows x 512 halves) ----
  for (int u = tid; u < 16 * 32; u += TPB) {
    const int r = u / 32, c = (u % 32) * 16;
    const uint4* s = (const uint4*)(wo16 + (size_t)(n0 + r) * HDIM + c);
    uint4* dv = (uint4*)&lds_wo[r * WOP + c];
    dv[0] = s[0];
    dv[1] = s[1];
  }
  __syncthreads();

  // Hoisted per-lane biases (one output column each)
  const float b_r  = b_ih[c0 + ln] + b_hh[c0 + ln];
  const float b_z  = b_ih[512 + c0 + ln] + b_hh[512 + c0 + ln];
  const float b_nx = b_ih[1024 + c0 + ln];
  const float b_nh = b_hh[1024 + c0 + ln];
  const float b_e  = b_o[n0 + ln];

  v8f acc_r[2], acc_z[2], acc_nx[2], acc_nh[2];

  // One MMA K-chunk: load B frags (3 gates) + A frags (2 M-tiles), 6 WMMAs.
  auto mma_chunk = [&](int kc, v8f(&accn)[2]) {
    const _Float16* abase = lds_a + (kc & 1) * ABUF;
    v16h bf0 = *(const v16h*)&lds_w[(0 * 16 + ln) * WP + kc * 32 + hifl * 16];
    v16h bf1 = *(const v16h*)&lds_w[(1 * 16 + ln) * WP + kc * 32 + hifl * 16];
    v16h bf2 = *(const v16h*)&lds_w[(2 * 16 + ln) * WP + kc * 32 + hifl * 16];
#pragma unroll
    for (int mt = 0; mt < 2; ++mt) {
      const int row = 32 * w + 16 * mt + ln;
      const v8h* pa = (const v8h*)&abase[row * AP + hifl * 16];
      union { v16h v; v8h p[2]; } ua;
      ua.p[0] = pa[0];   // K {0..7} or {16..23}
      ua.p[1] = pa[1];   // K {8..15} or {24..31}
      acc_r[mt] = wmma_f16(ua.v, bf0, acc_r[mt]);
      acc_z[mt] = wmma_f16(ua.v, bf1, acc_z[mt]);
      accn[mt]  = wmma_f16(ua.v, bf2, accn[mt]);
    }
  };

  for (int t = 0; t < S_LEN; ++t) {
    const _Float16* __restrict__ hcur  = (t & 1) ? hb1 : hb0;
    _Float16* __restrict__       hnext = (t & 1) ? hb0 : hb1;

    // Prefetch x[t] rows (used by the error projection after the barrier).
    __builtin_prefetch(x + ((size_t)t * BATCH_ + tid) * DDIM, 0, 1);

#pragma unroll
    for (int mt = 0; mt < 2; ++mt) {
      acc_r[mt]  = splat8(b_r);
      acc_z[mt]  = splat8(b_z);
      acc_nx[mt] = splat8(b_nx);
      acc_nh[mt] = splat8(b_nh);
    }

    // Stage K-chunk kc of A = [x[t-1] | h] into lds_a buffer (kc&1).
    // is_x (kc<4): f32 x rows, convert to f16 through VGPRs (ds_store).
    // else: f16 h rows, async DMA straight to LDS (ASYNCcnt).
    auto stage = [&](int kc, bool is_x) {
      const int row = tid;
      _Float16* dst = lds_a + (kc & 1) * ABUF + row * AP;
      if (is_x) {
        const float4* s =
            (const float4*)(x + ((size_t)(t - 1) * BATCH_ + row) * DDIM + kc * 32);
#pragma unroll
        for (int i = 0; i < 4; ++i) {
          const float4 f0 = s[2 * i], f1 = s[2 * i + 1];
          union { _Float16 h[8]; uint4 u; } pk;
          pk.h[0] = (_Float16)f0.x; pk.h[1] = (_Float16)f0.y;
          pk.h[2] = (_Float16)f0.z; pk.h[3] = (_Float16)f0.w;
          pk.h[4] = (_Float16)f1.x; pk.h[5] = (_Float16)f1.y;
          pk.h[6] = (_Float16)f1.z; pk.h[7] = (_Float16)f1.w;
          ((uint4*)dst)[i] = pk.u;
        }
      } else {
        const _Float16* gsrc = hcur + (size_t)row * HDIM + (kc - 4) * 32;
        async_copy64((unsigned)(uintptr_t)dst, gsrc);
      }
    };

    const int kc0 = (t == 0) ? 4 : 0;   // x_in is zero at t==0
    stage(kc0, kc0 < 4);

    // ---- x-part of gates (K-chunks 0..3), accumulates into acc_nx ----
    for (int kc = kc0; kc < 4; ++kc) {
      wait_async0();
      __syncthreads();
      stage(kc + 1, kc + 1 < 4);        // kc+1 <= 4, chunk 4 is the first h chunk
      mma_chunk(kc, acc_nx);
    }

    // ---- h-part of gates (K-chunks 4..19), accumulates into acc_nh ----
#pragma unroll 2
    for (int kc = 4; kc < 20; ++kc) {
      wait_async0();      // publish async LDS writes for buffer (kc&1)
      __syncthreads();
      if (kc + 1 < 20) stage(kc + 1, false);  // overlap next copy with WMMAs
      mma_chunk(kc, acc_nh);
    }

    // ---- Gate nonlinearities + state update for our 16 h-columns ----
#pragma unroll
    for (int mt = 0; mt < 2; ++mt) {
#pragma unroll
      for (int v = 0; v < 8; ++v) {
        const int m = 32 * w + 16 * mt + v + 8 * hifl;
        const float rr = sigf(acc_r[mt][v]);
        const float zz = sigf(acc_z[mt][v]);
        const float nn = tanhf(acc_nx[mt][v] + rr * acc_nh[mt][v]);
        const float hp = (float)hcur[(size_t)m * HDIM + c0 + ln];
        hnext[(size_t)m * HDIM + c0 + ln] =
            (_Float16)((1.0f - zz) * nn + zz * hp);
      }
    }

    grid_sync(cnt, (unsigned)(t + 1) * NBLK);

    // ---- err[t] = h_new @ w_o.T + b_o - x[t]  (waves 0..3) ----
    if (w < 4) {
      const int mtile = mgrp * 4 + w;
      const int rowA  = mtile * 16 + ln;
      v8f eacc = splat8(b_e);
#pragma unroll
      for (int kc = 0; kc < 16; ++kc) {
        const int k0 = kc * 32 + hifl * 16;
        union { v16h v; v8h p[2]; } ua;
        ua.p[0] = *(const v8h*)(hnext + (size_t)rowA * HDIM + k0);
        ua.p[1] = *(const v8h*)(hnext + (size_t)rowA * HDIM + k0 + 8);
        const v16h bw =
            *(const v16h*)&lds_wo[ln * WOP + kc * 32 + hifl * 16];
        eacc = wmma_f16(ua.v, bw, eacc);
      }
#pragma unroll
      for (int v = 0; v < 8; ++v) {
        const int m = mtile * 16 + v + 8 * hifl;
        const size_t o = ((size_t)t * BATCH_ + m) * DDIM + n0 + ln;
        out[o] = eacc[v] - x[o];
      }
    }
  }
}

extern "C" void kernel_launch(void* const* d_in, const int* in_sizes, int n_in,
                              void* d_out, int out_size, void* d_ws,
                              size_t ws_size, hipStream_t stream) {
  (void)in_sizes; (void)n_in; (void)out_size; (void)ws_size;
  const float* x      = (const float*)d_in[0];
  const float* h_init = (const float*)d_in[1];
  const float* w_ih   = (const float*)d_in[2];
  const float* w_hh   = (const float*)d_in[3];
  const float* b_ih   = (const float*)d_in[4];
  const float* b_hh   = (const float*)d_in[5];
  const float* w_o    = (const float*)d_in[6];
  const float* b_o    = (const float*)d_in[7];
  float* out = (float*)d_out;

  char* ws = (char*)d_ws;
  _Float16* wcat = (_Float16*)(ws);             // 1536*640*2 = 1,966,080 B
  _Float16* wo16 = (_Float16*)(ws + 1966080);   // 128*512*2  =   131,072 B
  _Float16* hb0  = (_Float16*)(ws + 2097152);   // 256*512*2  =   262,144 B
  _Float16* hb1  = (_Float16*)(ws + 2359296);   // 256*512*2  =   262,144 B
  unsigned* cnt  = (unsigned*)(ws + 2621440);   // barrier counter

  hipMemsetAsync(cnt, 0, 256, stream);
  gru_prep<<<(3 * HDIM * KTOT + TPB - 1) / TPB, TPB, 0, stream>>>(
      w_ih, w_hh, w_o, h_init, wcat, wo16, hb0);
  gru_persistent<<<NBLK, TPB, 0, stream>>>(x, b_ih, b_hh, b_o, wcat, wo16, hb0,
                                           hb1, out, cnt);
}
